// Dynamic_conv2d_90701119357043
// MI455X (gfx1250) — matrix operand order repaired and checked by reading, not compiled, not run on hardware
//
#include <hip/hip_runtime.h>
#include <hip/hip_bf16.h>

// Sizes fixed by the reference:
#define B_  16
#define C_  256
#define H_  64
#define W_  64
#define K_  4
#define O_  256
#define HID 65
#define TEMP 34.0f

typedef _Float16 half8 __attribute__((ext_vector_type(8)));
typedef _Float16 v16h  __attribute__((ext_vector_type(16)));
typedef float    v8f   __attribute__((ext_vector_type(8)));

union Vec16 { v16h v; half8 h[2]; };

// ---------------------------------------------------------------------------
// 1) Global average pool: one block per (b,c), 256 threads reduce 4096 elems.
// ---------------------------------------------------------------------------
__global__ __launch_bounds__(256) void pool_kernel(const float* __restrict__ x,
                                                   float* __restrict__ pooled) {
  const int bc  = blockIdx.x;               // b*C + c
  const int tid = threadIdx.x;
  const float4* p4 = (const float4*)(x + (size_t)bc * (H_ * W_));
  float s = 0.f;
  for (int i = tid; i < (H_ * W_) / 4; i += 256) {
    float4 v = p4[i];
    s += v.x + v.y + v.z + v.w;
  }
  __shared__ float red[256];
  red[tid] = s;
  __syncthreads();
  for (int off = 128; off > 0; off >>= 1) {
    if (tid < off) red[tid] += red[tid + off];
    __syncthreads();
  }
  if (tid == 0) pooled[bc] = red[0] * (1.f / (H_ * W_));
}

// ---------------------------------------------------------------------------
// 2) Attention MLP + softmax(T) + aggregated bias. Single tiny block.
// ---------------------------------------------------------------------------
__global__ __launch_bounds__(256) void att_kernel(const float* __restrict__ pooled,
                                                  const float* __restrict__ w_fc1,
                                                  const float* __restrict__ w_fc2,
                                                  const float* __restrict__ b_fc2,
                                                  const float* __restrict__ bias,
                                                  float* __restrict__ att,
                                                  float* __restrict__ agg_b) {
  __shared__ float att_s[B_ * K_];
  const int tid = threadIdx.x;
  if (tid < B_) {
    const int b = tid;
    float h[HID];
    for (int j = 0; j < HID; ++j) {
      float s = 0.f;
      for (int c = 0; c < C_; ++c) s += pooled[b * C_ + c] * w_fc1[j * C_ + c];
      h[j] = fmaxf(s, 0.f);
    }
    float lg[K_];
    float m = -1e30f;
    for (int k = 0; k < K_; ++k) {
      float s = b_fc2[k];
      for (int j = 0; j < HID; ++j) s += h[j] * w_fc2[k * HID + j];
      lg[k] = s * (1.0f / TEMP);
      m = fmaxf(m, lg[k]);
    }
    float e[K_], den = 0.f;
    for (int k = 0; k < K_; ++k) { e[k] = expf(lg[k] - m); den += e[k]; }
    for (int k = 0; k < K_; ++k) {
      const float a = e[k] / den;
      att_s[b * K_ + k] = a;
      att[b * K_ + k]   = a;
    }
  }
  __syncthreads();
  for (int i = tid; i < B_ * O_; i += 256) {
    const int b = i >> 8, o = i & (O_ - 1);
    float s = 0.f;
    for (int k = 0; k < K_; ++k) s += att_s[b * K_ + k] * bias[k * O_ + o];
    agg_b[i] = s;
  }
}

// ---------------------------------------------------------------------------
// 3) Aggregate per-sample conv kernels into f16 with WMMA-friendly layout:
//    wh[b][tap][o][c] (C contiguous so A-fragments are two b128 loads).
// ---------------------------------------------------------------------------
__global__ __launch_bounds__(256) void aggw_kernel(const float* __restrict__ weight,
                                                   const float* __restrict__ att,
                                                   _Float16* __restrict__ wh) {
  const int o = blockIdx.x;
  const int b = blockIdx.y;
  const int c = threadIdx.x;
  float acc[9] = {0.f, 0.f, 0.f, 0.f, 0.f, 0.f, 0.f, 0.f, 0.f};
  for (int k = 0; k < K_; ++k) {
    const float ak = att[b * K_ + k];
    const float* wp = weight + (((size_t)k * O_ + o) * C_ + c) * 9;
    for (int t = 0; t < 9; ++t) acc[t] += ak * wp[t];
  }
  for (int t = 0; t < 9; ++t)
    wh[(((size_t)(b * 9 + t)) * O_ + o) * C_ + c] = (_Float16)acc[t];
}

// ---------------------------------------------------------------------------
// 4) Implicit-GEMM conv via v_wmma_f32_16x16x32_f16.
//    Block = (b, output row y, 128-channel O-chunk); 8 waves, each wave owns
//    one 16-O M-tile and the FULL 64-pixel row: 4 accumulator tiles, so each
//    A fragment feeds 4 WMMAs (A traffic amortized 4x, no duplication).
//    K-loop: 8 chunks of 32 input channels x 9 taps -> 36 WMMA per chunk/wave.
//    x row slab (3 rows x 66 padded px x 32 ch, f16, transposed) staged in LDS
//    with div/mod-free indexing and fully-coalesced global reads.
// ---------------------------------------------------------------------------
#define CHS 32            // channels per K-chunk (== WMMA K)
#define PXP 66            // 64 + 2 halo columns

__global__ __launch_bounds__(256) void conv_kernel(const float* __restrict__ x,
                                                   const _Float16* __restrict__ wh,
                                                   const float* __restrict__ agg_b,
                                                   float* __restrict__ out) {
  __shared__ _Float16 xs[3 * PXP * CHS];     // 12672 B

  const int tid  = threadIdx.x;
  const int lane = tid & 31;
  const int wave = tid >> 5;                 // 8 M tiles of 16 O
  const int b    = blockIdx.z;
  const int y    = blockIdx.y;
  const int obase = blockIdx.x * 128 + wave * 16;
  const int n  = lane & 15;                  // tile column / A row
  const int hi = lane >> 4;                  // 0: lanes 0-15, 1: lanes 16-31

  // staging decomposition: 64 threads sweep interior columns, 4 groups x ch
  const int pp = tid & 63;                   // interior column (wc == pp)
  const int g  = tid >> 6;                   // 0..3 channel group

  v8f acc0 = {}, acc1 = {}, acc2 = {}, acc3 = {};

  for (int ct = 0; ct < C_ / CHS; ++ct) {
    const int cbase = ct * CHS;
    __syncthreads();                         // protect previous tile reads
    // Stage transposed f16 x-slab: xs[row][p][ch]; p=0 and p=65 are the
    // always-zero W halo (wc = -1 / 64); y halo rows are zero-filled.
    for (int row = 0; row < 3; ++row) {
      const int yc = y - 1 + row;
      const bool yok = (unsigned)yc < (unsigned)H_;
      const int yuse = yok ? yc : 0;
      for (int ch = g; ch < CHS; ch += 4) {
        const float* xp =
            x + (((size_t)b * C_ + cbase + ch) * H_ + yuse) * W_;
        const float v = yok ? xp[pp] : 0.f;
        xs[(row * PXP + (pp + 1)) * CHS + ch] = (_Float16)v;
        if (pp < 2)                          // halo columns p=0 and p=65
          xs[(row * PXP + pp * 65) * CHS + ch] = (_Float16)0.f;
      }
    }
    __syncthreads();

    if (ct + 1 < C_ / CHS)                   // hint next A chunk toward L2
      __builtin_prefetch(wh + (((size_t)(b * 9)) * O_ + obase + n) * C_ + cbase + CHS, 0, 1);

    for (int t = 0; t < 9; ++t) {
      const int r = t / 3, s = t - r * 3;
      // A fragment (16 O x 32 C), ISA 16-bit A layout:
      //   lanes 0-15: K 0..7 then 16..23 ; lanes 16-31: K 8..15 then 24..31
      const _Float16* ap =
          wh + (((size_t)(b * 9 + t)) * O_ + obase + n) * C_ + cbase + hi * 8;
      Vec16 a;
      a.h[0] = *(const half8*)(ap);
      a.h[1] = *(const half8*)(ap + 16);

      // B fragments (32 C x 16 px), same K order as A:
      //   lanes 0-15 -> K 0..7,16..23, lanes 16-31 -> K 8..15,24..31, column = lane%16.
      const _Float16* bp = &xs[(r * PXP + n + s) * CHS + hi * 8];
      Vec16 b0, b1, b2, b3;
      b0.h[0] = *(const half8*)(bp);
      b0.h[1] = *(const half8*)(bp + 16);
      b1.h[0] = *(const half8*)(bp + 16 * CHS);
      b1.h[1] = *(const half8*)(bp + 16 * CHS + 16);
      b2.h[0] = *(const half8*)(bp + 32 * CHS);
      b2.h[1] = *(const half8*)(bp + 32 * CHS + 16);
      b3.h[0] = *(const half8*)(bp + 48 * CHS);
      b3.h[1] = *(const half8*)(bp + 48 * CHS + 16);

      acc0 = __builtin_amdgcn_wmma_f32_16x16x32_f16(false, a.v, false, b0.v,
                                                    (short)0, acc0, false, false);
      acc1 = __builtin_amdgcn_wmma_f32_16x16x32_f16(false, a.v, false, b1.v,
                                                    (short)0, acc1, false, false);
      acc2 = __builtin_amdgcn_wmma_f32_16x16x32_f16(false, a.v, false, b2.v,
                                                    (short)0, acc2, false, false);
      acc3 = __builtin_amdgcn_wmma_f32_16x16x32_f16(false, a.v, false, b3.v,
                                                    (short)0, acc3, false, false);
    }
  }

  // Epilogue: C/D layout — VGPR i: M = i + 8*hi, N = lane%16. Add agg bias.
  for (int i = 0; i < 8; ++i) {
    const int o = obase + i + hi * 8;
    const float bb = agg_b[b * O_ + o];
    const size_t base = (((size_t)b * O_ + o) * H_ + y) * W_;
    out[base +  0 + n] = acc0[i] + bb;
    out[base + 16 + n] = acc1[i] + bb;
    out[base + 32 + n] = acc2[i] + bb;
    out[base + 48 + n] = acc3[i] + bb;
  }
}

// ---------------------------------------------------------------------------
// Launch. Workspace layout (bytes):
//   [0)        pooled  B*C f32           = 16384
//   [16384)    att     B*K f32           = 256   (padded to 1024)
//   [17408)    agg_b   B*O f32           = 16384
//   [65536)    wh      B*9*O*C f16       = 18,874,368
// Total ~18.94 MB.
// ---------------------------------------------------------------------------
extern "C" void kernel_launch(void* const* d_in, const int* in_sizes, int n_in,
                              void* d_out, int out_size, void* d_ws, size_t ws_size,
                              hipStream_t stream) {
  const float* x      = (const float*)d_in[0];
  const float* w_fc1  = (const float*)d_in[1];
  const float* w_fc2  = (const float*)d_in[2];
  const float* b_fc2  = (const float*)d_in[3];
  const float* weight = (const float*)d_in[4];
  const float* bias   = (const float*)d_in[5];
  float* out = (float*)d_out;

  char* ws = (char*)d_ws;
  float*    pooled = (float*)(ws);
  float*    att    = (float*)(ws + 16384);
  float*    agg_b  = (float*)(ws + 17408);
  _Float16* wh     = (_Float16*)(ws + 65536);

  pool_kernel<<<B_ * C_, 256, 0, stream>>>(x, pooled);
  att_kernel<<<1, 256, 0, stream>>>(pooled, w_fc1, w_fc2, b_fc2, bias, att, agg_b);
  aggw_kernel<<<dim3(O_, B_), C_, 0, stream>>>(weight, att, wh);
  conv_kernel<<<dim3(O_ / 128, H_, B_), 256, 0, stream>>>(x, wh, agg_b, out);
}
